// CEBlock_84155589198434
// MI455X (gfx1250) — matrix operand order repaired and checked by reading, not compiled, not run on hardware
//
#include <hip/hip_runtime.h>
#include <stdint.h>

// ---------------------------------------------------------------------------
// Types for CDNA5 WMMA (wave32): D(16x16 f32) = A(16x32 bf16) x B(32x16 bf16) + C
// ---------------------------------------------------------------------------
typedef __attribute__((ext_vector_type(16))) __bf16 bf16x16;
typedef __attribute__((ext_vector_type(8)))  float  f32x8;
typedef __attribute__((ext_vector_type(4)))  unsigned int v4u;
typedef __attribute__((ext_vector_type(8)))  int v8i;
typedef __attribute__((ext_vector_type(4)))  int v4i;
typedef __attribute__((address_space(3)))    uint16_t lds_u16;

__device__ __forceinline__ uint16_t f2bf(float f) {
  uint32_t u = __float_as_uint(f);
  uint32_t r = ((u >> 16) & 1u) + 0x7FFFu;   // round-to-nearest-even
  return (uint16_t)((u + r) >> 16);
}
__device__ __forceinline__ float bf2f(uint16_t h) {
  return __uint_as_float(((uint32_t)h) << 16);
}

union FragU { bf16x16 v; uint32_t u[8]; };

// A fragment: 16x32 bf16, row-major source with leading dim lda (elements).
// ISA 7.12.2: lanes0-15 M=lane, VGPR v<4 K=2v,2v+1 ; v>=4 K=16+2(v-4),..
// lanes16-31 same M, K offset +8 within each 16-K half.
__device__ __forceinline__ bf16x16 load_frag_a(const uint16_t* base, int lda, int lane) {
  FragU f;
  int m = lane & 15, hi = lane >> 4;
  const uint16_t* rp = base + (size_t)m * lda + hi * 8;
#pragma unroll
  for (int v = 0; v < 8; ++v) {
    int k = (v < 4) ? (2 * v) : (16 + 2 * (v - 4));
    f.u[v] = *reinterpret_cast<const uint32_t*>(rp + k);
  }
  return f.v;
}
// B fragment from transposed operand Bt[N][K] row-major (ldb elements):
// lanes0-15 N=lane, same K order as A: v<4 K=2v,2v+1 ; v>=4 K=16+2(v-4),..; lanes16-31 K +8.
__device__ __forceinline__ bf16x16 load_frag_b(const uint16_t* base, int ldb, int lane) {
  FragU f;
  int n = lane & 15, hi = lane >> 4;
  const uint16_t* rp = base + (size_t)n * ldb + hi * 8;
#pragma unroll
  for (int v = 0; v < 8; ++v)
    f.u[v] = *reinterpret_cast<const uint32_t*>(rp + ((v < 4) ? (2 * v) : (16 + 2 * (v - 4))));
  return f.v;
}

// ---------------------------------------------------------------------------
// Weight transpose + f32 -> bf16 convert:  wt[n*K + k] = bf16(w[k*N + n])
// ---------------------------------------------------------------------------
__global__ void convT_kernel(const float* __restrict__ w, uint16_t* __restrict__ wt,
                             int K, int N) {
  long idx = (long)blockIdx.x * blockDim.x + threadIdx.x;
  if (idx >= (long)K * N) return;
  int n = (int)(idx % N);
  int k = (int)(idx / N);
  wt[(size_t)n * K + k] = f2bf(w[idx]);
}

// ---------------------------------------------------------------------------
// LayerNorm over C=512, one row per block (256 threads), output bf16
// ---------------------------------------------------------------------------
__global__ void ln512_kernel(const float* __restrict__ x, const float* __restrict__ g,
                             const float* __restrict__ b, uint16_t* __restrict__ out) {
  const int C = 512;
  int row = blockIdx.x;
  const float* xr = x + (size_t)row * C;
  float v0 = xr[threadIdx.x], v1 = xr[threadIdx.x + 256];
  __shared__ float s1[256], s2[256];
  s1[threadIdx.x] = v0 + v1;
  s2[threadIdx.x] = v0 * v0 + v1 * v1;
  __syncthreads();
  for (int off = 128; off > 0; off >>= 1) {
    if (threadIdx.x < off) {
      s1[threadIdx.x] += s1[threadIdx.x + off];
      s2[threadIdx.x] += s2[threadIdx.x + off];
    }
    __syncthreads();
  }
  float mean = s1[0] * (1.0f / C);
  float var  = s2[0] * (1.0f / C) - mean * mean;
  float inv  = rsqrtf(var + 1e-5f);
  uint16_t* orow = out + (size_t)row * C;
  orow[threadIdx.x]       = f2bf((v0 - mean) * inv * g[threadIdx.x]       + b[threadIdx.x]);
  orow[threadIdx.x + 256] = f2bf((v1 - mean) * inv * g[threadIdx.x + 256] + b[threadIdx.x + 256]);
}

// ---------------------------------------------------------------------------
// Generic bf16 WMMA GEMM: out = act(A[M,K] @ Bt[N,K]^T + bias) (+res) (+s*extra)
// One wave per 16x64 output strip: the A fragment is reused by 4 B fragments
// (4 WMMAs per A load) to cut A-side traffic 4x. 256 threads = 8 waves/block.
// act: 0 = none, 1 = relu
// ---------------------------------------------------------------------------
__global__ __launch_bounds__(256)
void gemm_bf16_kernel(const uint16_t* __restrict__ A, int lda,
                      const uint16_t* __restrict__ Bt, int ldb,
                      const float* __restrict__ bias,
                      const float* __restrict__ residual,
                      const float* __restrict__ extra, float extraScale,
                      float* __restrict__ outF, uint16_t* __restrict__ outB,
                      int ldo, int M, int N, int K, int act) {
  int wave = threadIdx.x >> 5;
  int lane = threadIdx.x & 31;
  int strip = blockIdx.x * (blockDim.x >> 5) + wave;
  int nstrips = N >> 6;
  int tm = strip / nstrips, tn = strip % nstrips;
  if (tm * 16 >= M) return;   // wave-uniform, EXEC stays all-ones for WMMA

  f32x8 acc[4];
  f32x8 zero = {};
#pragma unroll
  for (int i = 0; i < 4; ++i) acc[i] = zero;

  const uint16_t* arow = A  + (size_t)tm * 16 * lda;
  const uint16_t* brow = Bt + (size_t)tn * 64 * ldb;
  for (int k0 = 0; k0 < K; k0 += 32) {
    __builtin_prefetch(arow + k0 + 256, 0, 0);   // global_prefetch_b8
    bf16x16 a = load_frag_a(arow + k0, lda, lane);
#pragma unroll
    for (int nt = 0; nt < 4; ++nt) {
      bf16x16 bb = load_frag_b(brow + (size_t)nt * 16 * ldb + k0, ldb, lane);
      acc[nt] = __builtin_amdgcn_wmma_f32_16x16x32_bf16(false, a, false, bb,
                                                        (short)0, acc[nt], false, false);
    }
  }
  int lx = lane & 15;
  int hi = lane >> 4;
#pragma unroll
  for (int nt = 0; nt < 4; ++nt) {
    int n = tn * 64 + nt * 16 + lx;
    float bv = bias ? bias[n] : 0.0f;
#pragma unroll
    for (int r = 0; r < 8; ++r) {
      int m = tm * 16 + r + hi * 8;
      float v = acc[nt][r] + bv;
      if (act == 1) v = fmaxf(v, 0.0f);
      size_t idx = (size_t)m * ldo + n;
      if (residual) v += residual[idx];
      if (extra)    v += extraScale * extra[idx];
      if (outF) outF[idx] = v;
      if (outB) outB[idx] = f2bf(v);
    }
  }
}

// ---------------------------------------------------------------------------
// Flash attention, hd=64, 8 heads. Block = 128 threads (4 waves), 64 query
// rows per block, streaming K/V tiles of 64 keys through LDS with online
// softmax in f32. The K tile is fetched with the Tensor Data Mover
// (tensor_load_to_lds + s_wait_tensorcnt) when the builtin is available:
// D# describes a 2D tile 64x64 x bf16 with tensor_dim0_stride = 2C.
// V is staged transposed so it can feed the B-operand of the P@V WMMA.
// ---------------------------------------------------------------------------
__global__ __launch_bounds__(128)
void attn_kernel(const uint16_t* __restrict__ qg, const uint16_t* __restrict__ kvg,
                 uint16_t* __restrict__ og, int N, int C, int nheads) {
  const int hd = 64;
  int qtiles = N / 64;
  int qt  = blockIdx.x % qtiles;
  int tmp = blockIdx.x / qtiles;
  int h = tmp % nheads;
  int b = tmp / nheads;
  int wave = threadIdx.x >> 5, lane = threadIdx.x & 31;
  int hi = lane >> 4, lx = lane & 15;

  __shared__ uint16_t qs[64 * 64];
  __shared__ uint16_t ks[64 * 64];
  __shared__ uint16_t vts[64 * 64];      // V transposed: vts[d*64 + key]
  __shared__ uint16_t ps[4][16 * 64];    // per-wave P tile

  // load Q tile [64 x 64]
  size_t qbase = ((size_t)(b * N + qt * 64)) * C + h * hd;
  for (int i = threadIdx.x; i < 64 * 64; i += blockDim.x) {
    int r = i >> 6, d = i & 63;
    qs[i] = qg[qbase + (size_t)r * C + d];
  }

  f32x8 zero = {};
  f32x8 oacc[4];
#pragma unroll
  for (int i = 0; i < 4; ++i) oacc[i] = zero;
  float mrow[8], lrow[8];
#pragma unroll
  for (int r = 0; r < 8; ++r) { mrow[r] = -INFINITY; lrow[r] = 0.0f; }

  size_t kbase = ((size_t)(b * N)) * (2 * C) + h * hd;
  size_t vbase = kbase + C;
  const uint16_t* qw = qs + wave * 16 * 64;
  uint16_t* pw = ps[wave];

#if __has_builtin(__builtin_amdgcn_tensor_load_to_lds)
  // LDS byte offset of the K staging tile (generic -> LDS addrspace cast)
  uint32_t ks_lds = (uint32_t)(uintptr_t)(lds_u16*)ks;
#endif

  for (int jt = 0; jt < qtiles; ++jt) {
    __syncthreads();   // protect ks/vts before overwrite

#if __has_builtin(__builtin_amdgcn_tensor_load_to_lds)
    if (wave == 0) {
      // TDM: 2D tile, 64 rows x 64 bf16, row stride 2C elements, into ks[]
      uint64_t ga = (uint64_t)(uintptr_t)(kvg + kbase + (size_t)(jt * 64) * (2 * C));
      v4u g0;
      g0[0] = 1u;                                       // count=1, user desc
      g0[1] = ks_lds;                                   // lds_addr (bytes)
      g0[2] = (uint32_t)ga;                             // global_addr[31:0]
      g0[3] = ((uint32_t)(ga >> 32) & 0x01FFFFFFu) | 0x80000000u;  // [56:32] | type=2
      v8i g1;
      g1[0] = 0x00010000;                               // data_size=1 (2 bytes)
      g1[1] = (64 << 16);                               // tensor_dim0 = 64 (lo16 @bits63:48)
      g1[2] = (64 << 16);                               // tensor_dim1 = 64 (lo16 @bits95:80)
      g1[3] = (64 << 16);                               // tile_dim0   = 64 (bits127:112)
      g1[4] = 64;                                       // tile_dim1=64, tile_dim2=0
      g1[5] = 2 * C;                                    // tensor_dim0_stride
      g1[6] = 0;
      g1[7] = 0;
      v4i gz = {};
      v8i gz8 = {};
      // 6-arg form (clang-23 / therock headers): groups 0..3 + spare group + cpol
      __builtin_amdgcn_tensor_load_to_lds(g0, g1, gz, gz, gz8, 0);
      __builtin_amdgcn_s_wait_tensorcnt(0);
    }
    // V (transposed) cooperative load
    for (int i = threadIdx.x; i < 64 * 64; i += blockDim.x) {
      int r = i >> 6, d = i & 63;
      size_t row = (size_t)(jt * 64 + r) * (2 * C);
      vts[d * 64 + r] = kvg[vbase + row + d];
    }
#else
    for (int i = threadIdx.x; i < 64 * 64; i += blockDim.x) {
      int r = i >> 6, d = i & 63;
      size_t row = (size_t)(jt * 64 + r) * (2 * C);
      ks[i]           = kvg[kbase + row + d];
      vts[d * 64 + r] = kvg[vbase + row + d];
    }
#endif
    __syncthreads();

    // S = (Q @ K^T) * hd^-0.5 : 4 n-tiles x 2 k-steps
    float sfr[4][8];
#pragma unroll
    for (int nt = 0; nt < 4; ++nt) {
      f32x8 c = zero;
#pragma unroll
      for (int kk = 0; kk < 64; kk += 32) {
        bf16x16 a  = load_frag_a(qw + kk, 64, lane);
        bf16x16 bb = load_frag_b(ks + nt * 16 * 64 + kk, 64, lane);
        c = __builtin_amdgcn_wmma_f32_16x16x32_bf16(false, a, false, bb,
                                                    (short)0, c, false, false);
      }
#pragma unroll
      for (int r = 0; r < 8; ++r) sfr[nt][r] = c[r] * 0.125f;
    }

    // online softmax update (butterfly within 16-lane half = one output row)
    float corr[8];
#pragma unroll
    for (int r = 0; r < 8; ++r) {
      float t = fmaxf(fmaxf(sfr[0][r], sfr[1][r]), fmaxf(sfr[2][r], sfr[3][r]));
#pragma unroll
      for (int msk = 8; msk >= 1; msk >>= 1) t = fmaxf(t, __shfl_xor(t, msk, 32));
      float newm = fmaxf(mrow[r], t);
      corr[r] = __expf(mrow[r] - newm);
      mrow[r] = newm;
      float rs = 0.0f;
#pragma unroll
      for (int nt = 0; nt < 4; ++nt) {
        float p = __expf(sfr[nt][r] - newm);
        sfr[nt][r] = p;
        rs += p;
      }
#pragma unroll
      for (int msk = 8; msk >= 1; msk >>= 1) rs += __shfl_xor(rs, msk, 32);
      lrow[r] = lrow[r] * corr[r] + rs;
    }
#pragma unroll
    for (int nt = 0; nt < 4; ++nt)
#pragma unroll
      for (int r = 0; r < 8; ++r) oacc[nt][r] *= corr[r];

    // P (C-layout f32) -> LDS bf16 row-major [16 x 64], re-read as A-fragments
#pragma unroll
    for (int nt = 0; nt < 4; ++nt)
#pragma unroll
      for (int r = 0; r < 8; ++r)
        pw[(r + hi * 8) * 64 + nt * 16 + lx] = f2bf(sfr[nt][r]);

    // O += P @ V
#pragma unroll
    for (int ht = 0; ht < 4; ++ht) {
      f32x8 c = oacc[ht];
#pragma unroll
      for (int kk = 0; kk < 64; kk += 32) {
        bf16x16 a  = load_frag_a(pw + kk, 64, lane);
        bf16x16 bb = load_frag_b(vts + ht * 16 * 64 + kk, 64, lane);
        c = __builtin_amdgcn_wmma_f32_16x16x32_bf16(false, a, false, bb,
                                                    (short)0, c, false, false);
      }
      oacc[ht] = c;
    }
  }

  // write O / l  (bf16, laid out [B*N, C] at head column)
  size_t obase = ((size_t)(b * N + qt * 64 + wave * 16)) * C + h * hd;
#pragma unroll
  for (int ht = 0; ht < 4; ++ht)
#pragma unroll
    for (int r = 0; r < 8; ++r) {
      int row = r + hi * 8, col = ht * 16 + lx;
      og[obase + (size_t)row * C + col] = f2bf(oacc[ht][r] / lrow[r]);
    }
}

// ---------------------------------------------------------------------------
// Depthwise 3x3 conv (pad 1) + bias + exact GELU. Channels contiguous ->
// coalesced. H,W read from device scalars (graph-capture safe).
// ---------------------------------------------------------------------------
__global__ void dwconv_gelu_kernel(const uint16_t* __restrict__ hin,
                                   const float* __restrict__ k9,
                                   const float* __restrict__ kb,
                                   uint16_t* __restrict__ out,
                                   const int* __restrict__ Hp,
                                   const int* __restrict__ Wp,
                                   int Cm, long total) {
  long idx = (long)blockIdx.x * blockDim.x + threadIdx.x;
  if (idx >= total) return;
  int H = *Hp, W = *Wp;
  int c = (int)(idx % Cm);
  long s = idx / Cm;
  int xw = (int)(s % W); s /= W;
  int y  = (int)(s % H);
  long b = s / H;
  float acc = kb[c];
#pragma unroll
  for (int ky = -1; ky <= 1; ++ky) {
    int yy = y + ky;
    if (yy < 0 || yy >= H) continue;
#pragma unroll
    for (int kx = -1; kx <= 1; ++kx) {
      int xx = xw + kx;
      if (xx < 0 || xx >= W) continue;
      float wgt = k9[(size_t)c * 9 + (ky + 1) * 3 + (kx + 1)];
      float val = bf2f(hin[(((size_t)(b * H + yy)) * W + xx) * Cm + c]);
      acc += wgt * val;
    }
  }
  float g = 0.5f * acc * (1.0f + erff(acc * 0.70710678118654752f));
  out[idx] = f2bf(g);
}

// ---------------------------------------------------------------------------
// kernel_launch
// ---------------------------------------------------------------------------
extern "C" void kernel_launch(void* const* d_in, const int* in_sizes, int n_in,
                              void* d_out, int out_size, void* d_ws, size_t ws_size,
                              hipStream_t stream) {
  const float* x     = (const float*)d_in[0];
  const float* g1    = (const float*)d_in[1];
  const float* b1    = (const float*)d_in[2];
  const float* wq    = (const float*)d_in[3];
  const float* bq    = (const float*)d_in[4];
  const float* wkv   = (const float*)d_in[5];
  const float* bkv   = (const float*)d_in[6];
  const float* wproj = (const float*)d_in[7];
  const float* bproj = (const float*)d_in[8];
  const float* g2    = (const float*)d_in[9];
  const float* b2    = (const float*)d_in[10];
  const float* wfc1  = (const float*)d_in[11];
  const float* bfc1  = (const float*)d_in[12];
  const float* dwk   = (const float*)d_in[13];
  const float* dwb   = (const float*)d_in[14];
  const float* wfc2  = (const float*)d_in[15];
  const float* bfc2  = (const float*)d_in[16];
  const float* wa1   = (const float*)d_in[17];
  const float* ba1   = (const float*)d_in[18];
  const float* wa2   = (const float*)d_in[19];
  const float* ba2   = (const float*)d_in[20];
  const int*   Hp    = (const int*)d_in[21];
  const int*   Wp    = (const int*)d_in[22];

  const int C = 512, Cm = 2048, Ca = 128, Bn = 2, Hh = 48, Ww = 48, HEADS = 8;
  const int N = Hh * Ww;          // 2304
  const int BN = Bn * N;          // 4608

  // workspace carve-out
  char* wsp = (char*)d_ws;
  size_t off = 0;
  auto carve = [&](size_t bytes) -> void* {
    void* p = wsp + off;
    off += (bytes + 255) & ~(size_t)255;
    return p;
  };
  uint16_t* xn_bf   = (uint16_t*)carve((size_t)BN * C * 2);
  uint16_t* wqT     = (uint16_t*)carve((size_t)C * C * 2);
  uint16_t* wkvT    = (uint16_t*)carve((size_t)C * 2 * C * 2);
  uint16_t* wprojT  = (uint16_t*)carve((size_t)C * C * 2);
  uint16_t* wfc1T   = (uint16_t*)carve((size_t)C * Cm * 2);
  uint16_t* wfc2T   = (uint16_t*)carve((size_t)Cm * C * 2);
  uint16_t* wa1T    = (uint16_t*)carve((size_t)C * Ca * 2);
  uint16_t* wa2T    = (uint16_t*)carve((size_t)Ca * C * 2);
  uint16_t* q_bf    = (uint16_t*)carve((size_t)BN * C * 2);
  uint16_t* kv_bf   = (uint16_t*)carve((size_t)BN * 2 * C * 2);
  uint16_t* o_bf    = (uint16_t*)carve((size_t)BN * C * 2);
  float*    x1_f    = (float*)   carve((size_t)BN * C * 4);
  uint16_t* x1_bf   = (uint16_t*)carve((size_t)BN * C * 2);
  uint16_t* xn2_bf  = (uint16_t*)carve((size_t)BN * C * 2);
  uint16_t* h_bf    = (uint16_t*)carve((size_t)BN * Cm * 2);
  uint16_t* hh_bf   = (uint16_t*)carve((size_t)BN * Cm * 2);
  uint16_t* a1_bf   = (uint16_t*)carve((size_t)BN * Ca * 2);
  float*    a_f     = (float*)   carve((size_t)BN * C * 4);

  auto ceil_div = [](long a, long b) { return (int)((a + b - 1) / b); };

  // 1) weight transpose + bf16 convert
  convT_kernel<<<ceil_div((long)C * C, 256), 256, 0, stream>>>(wq, wqT, C, C);
  convT_kernel<<<ceil_div((long)C * 2 * C, 256), 256, 0, stream>>>(wkv, wkvT, C, 2 * C);
  convT_kernel<<<ceil_div((long)C * C, 256), 256, 0, stream>>>(wproj, wprojT, C, C);
  convT_kernel<<<ceil_div((long)C * Cm, 256), 256, 0, stream>>>(wfc1, wfc1T, C, Cm);
  convT_kernel<<<ceil_div((long)Cm * C, 256), 256, 0, stream>>>(wfc2, wfc2T, Cm, C);
  convT_kernel<<<ceil_div((long)C * Ca, 256), 256, 0, stream>>>(wa1, wa1T, C, Ca);
  convT_kernel<<<ceil_div((long)Ca * C, 256), 256, 0, stream>>>(wa2, wa2T, Ca, C);

  // 2) LN1
  ln512_kernel<<<BN, 256, 0, stream>>>(x, g1, b1, xn_bf);

  // one wave per 16x64 strip, 8 waves per block
  auto gemm_blocks = [](int M, int Ncols) { return ((M / 16) * (Ncols / 64)) / 8; };

  // 3) q = xn @ wq + bq
  gemm_bf16_kernel<<<gemm_blocks(BN, C), 256, 0, stream>>>(
      xn_bf, C, wqT, C, bq, nullptr, nullptr, 0.0f,
      nullptr, q_bf, C, BN, C, C, 0);
  // 4) kv = xn @ wkv + bkv
  gemm_bf16_kernel<<<gemm_blocks(BN, 2 * C), 256, 0, stream>>>(
      xn_bf, C, wkvT, C, bkv, nullptr, nullptr, 0.0f,
      nullptr, kv_bf, 2 * C, BN, 2 * C, C, 0);

  // 5) flash attention
  attn_kernel<<<Bn * HEADS * (N / 64), 128, 0, stream>>>(q_bf, kv_bf, o_bf, N, C, HEADS);

  // 6) x1 = x + o @ wproj + bproj  (f32 + bf16 copies)
  gemm_bf16_kernel<<<gemm_blocks(BN, C), 256, 0, stream>>>(
      o_bf, C, wprojT, C, bproj, x, nullptr, 0.0f,
      x1_f, x1_bf, C, BN, C, C, 0);

  // 7) LN2
  ln512_kernel<<<BN, 256, 0, stream>>>(x1_f, g2, b2, xn2_bf);

  // 8) h = xn2 @ w_fc1 + b_fc1
  gemm_bf16_kernel<<<gemm_blocks(BN, Cm), 256, 0, stream>>>(
      xn2_bf, C, wfc1T, C, bfc1, nullptr, nullptr, 0.0f,
      nullptr, h_bf, Cm, BN, Cm, C, 0);

  // 9) hh = gelu(dwconv(h) + dw_b)
  {
    long total = (long)BN * Cm;
    dwconv_gelu_kernel<<<ceil_div(total, 256), 256, 0, stream>>>(
        h_bf, dwk, dwb, hh_bf, Hp, Wp, Cm, total);
  }

  // 10) adapter: a1 = relu(x1 @ wa1 + ba1)
  gemm_bf16_kernel<<<gemm_blocks(BN, Ca), 256, 0, stream>>>(
      x1_bf, C, wa1T, C, ba1, nullptr, nullptr, 0.0f,
      nullptr, a1_bf, Ca, BN, Ca, C, 1);
  // 11) a = a1 @ wa2 + ba2  (f32)
  gemm_bf16_kernel<<<gemm_blocks(BN, C), 256, 0, stream>>>(
      a1_bf, Ca, wa2T, Ca, ba2, nullptr, nullptr, 0.0f,
      a_f, nullptr, C, BN, C, Ca, 0);

  // 12) out = x1 + (hh @ w_fc2 + b_fc2) + 0.5 * a
  gemm_bf16_kernel<<<gemm_blocks(BN, C), 256, 0, stream>>>(
      hh_bf, Cm, wfc2T, Cm, bfc2, x1_f, a_f, 0.5f,
      (float*)d_out, nullptr, C, BN, C, Cm, 0);
}
